// GCN_73186242723952
// MI455X (gfx1250) — compile-verified, matrix operand order checked by reading, not run on hardware
//
#include <hip/hip_runtime.h>

// ---- types -----------------------------------------------------------------
typedef __attribute__((ext_vector_type(16))) __bf16 v16bf;
typedef __attribute__((ext_vector_type(8)))  float  v8f;
typedef __attribute__((ext_vector_type(8)))  __bf16 bf16x8;
typedef __attribute__((ext_vector_type(4)))  float  f32x4;

#define NN   8192
#define DIMM 128
#define KSTEPS (NN / 32)
#define LROW 40                    // padded LDS row stride (bf16): 80B = 20 DW
                                   // => lane n starts at bank (n*20)%64: all 16
                                   // starts distinct, 4 banks apart -> b128
                                   // reads are conflict-free
#define LBUF (DIMM * LROW)         // 5120 bf16 = 10240 B per buffer

// ---------------------------------------------------------------------------
// Kernel 1: XW_T[n][k] = sum_c X[k][c] * W[c][n]   (bf16, stored transposed)
// ---------------------------------------------------------------------------
__global__ __launch_bounds__(32) void gcn_xw_kernel(
    const float* __restrict__ X, const float* __restrict__ W,
    __bf16* __restrict__ xwT) {
  const int lane = threadIdx.x & 31;
  const int l16  = lane & 15;
  const int half = lane >> 4;
  const int kbase = blockIdx.x * 16;   // rows of X  (K of the big GEMM)
  const int nbase = blockIdx.y * 16;   // cols of W  (N of the big GEMM)

  v8f acc = {};

#pragma unroll
  for (int c0 = 0; c0 < DIMM; c0 += 32) {
    const float* ap = X + (size_t)(kbase + l16) * DIMM + c0 + half * 8;
    f32x4 a0 = *(const f32x4*)(ap);
    f32x4 a1 = *(const f32x4*)(ap + 4);
    f32x4 a2 = *(const f32x4*)(ap + 16);
    f32x4 a3 = *(const f32x4*)(ap + 20);
    v16bf aF, bF;
#pragma unroll
    for (int j = 0; j < 4; ++j) {
      aF[j]      = (__bf16)a0[j];
      aF[4 + j]  = (__bf16)a1[j];
      aF[8 + j]  = (__bf16)a2[j];
      aF[12 + j] = (__bf16)a3[j];
    }
#pragma unroll
    for (int j = 0; j < 8; ++j) {
      bF[j]     = (__bf16)W[(size_t)(c0 + half * 8 + j) * DIMM + nbase + l16];
      bF[8 + j] = (__bf16)W[(size_t)(c0 + 16 + half * 8 + j) * DIMM + nbase + l16];
    }
    acc = __builtin_amdgcn_wmma_f32_16x16x32_bf16(false, aF, false, bF,
                                                  (short)0, acc, false, false);
  }

#pragma unroll
  for (int r = 0; r < 8; ++r) {
    const int m = r + half * 8;
    xwT[(size_t)(nbase + l16) * NN + kbase + m] = (__bf16)acc[r];
  }
}

// ---------------------------------------------------------------------------
// Kernel 2: out[m][n] = (adj[m][:] @ XW)[n] / rowsum(adj[m][:])
//
// Workgroup = 4 waves, each owning one 16-row M-tile (64 rows/WG).
// Per K-step the 8 KB B block (XW_T[:, k0:k0+32]) is staged into LDS once per
// WG via global_load_async_to_lds_b128 (ASYNCcnt, double-buffered, padded
// rows for conflict-free ds_load_b128) and shared by all 4 waves.
// adj is streamed once with nontemporal b128 loads; rowsum is fused as a 9th
// WMMA against an all-ones B fragment (same lane/VGPR slots as C => per-lane
// divide at the end).
// ---------------------------------------------------------------------------
__global__ __launch_bounds__(128) void gcn_agg_kernel(
    const float* __restrict__ adj, const __bf16* __restrict__ xwT,
    float* __restrict__ out) {
  // [buf][n][k(padded)] : 2 x 128 x 40 bf16 = 20 KB
  __shared__ __bf16 smem[2][DIMM][LROW];

  const int tid  = threadIdx.x;
  const int wave = tid >> 5;
  const int lane = tid & 31;
  const int l16  = lane & 15;
  const int half = lane >> 4;
  const int mbase = (blockIdx.x * 4 + wave) * 16;

  // --- async-copy assignment: 512 chunks of 16B (8KB payload); chunk c ->
  //     row n = c>>2, quarter q = c&3.  LDS offset = n*80 + q*16 (+buf*LBUF*2)
  const unsigned ldsBase = (unsigned)(size_t)(&smem[0][0][0]);
  unsigned    ldsOff[4];
  const char* gsrc[4];
#pragma unroll
  for (int j = 0; j < 4; ++j) {
    const int c = (wave * 4 + j) * 32 + lane;          // 0..511
    const int n = c >> 2;
    const int q = c & 3;
    ldsOff[j] = ldsBase + (unsigned)(n * (LROW * 2) + q * 16);
    gsrc[j]   = (const char*)xwT + (size_t)n * (NN * 2) + q * 16;
  }

  const v8f vzero = {};
  v8f acc[8];
#pragma unroll
  for (int t = 0; t < 8; ++t) acc[t] = vzero;
  v8f accSum = vzero;

  v16bf bOnes;
#pragma unroll
  for (int i = 0; i < 16; ++i) bOnes[i] = (__bf16)1.0f;

  const float* aRow = adj + (size_t)(mbase + l16) * NN + half * 8;

  // prologue: stage buffer 0 with k0 = 0
#pragma unroll
  for (int j = 0; j < 4; ++j)
    asm volatile("global_load_async_to_lds_b128 %0, %1, off"
                 :: "v"(ldsOff[j]), "v"(gsrc[j]) : "memory");

  for (int it = 0; it < KSTEPS; ++it) {
    // my async loads for the current buffer are done; barrier makes them
    // visible WG-wide and proves all readers of the other buffer finished.
    asm volatile("s_wait_asynccnt 0x0" ::: "memory");
    __syncthreads();

    if (it + 1 < KSTEPS) {                    // stage next buffer (overlaps)
      const unsigned bufAdd = (unsigned)(((it + 1) & 1) * (LBUF * 2));
      const int      kByte  = (it + 1) * 64;  // 32 bf16 = 64B per n-row
#pragma unroll
      for (int j = 0; j < 4; ++j)
        asm volatile("global_load_async_to_lds_b128 %0, %1, off"
                     :: "v"(ldsOff[j] + bufAdd), "v"(gsrc[j] + kByte)
                     : "memory");
    }

    const int cur = it & 1;
    const int k0  = it * 32;

    // A: stream adj once (nontemporal), prefetch ~1KB ahead
    __builtin_prefetch(aRow + k0 + 256, 0, 1);
    f32x4 a0 = __builtin_nontemporal_load((const f32x4*)(aRow + k0));
    f32x4 a1 = __builtin_nontemporal_load((const f32x4*)(aRow + k0 + 4));
    f32x4 a2 = __builtin_nontemporal_load((const f32x4*)(aRow + k0 + 16));
    f32x4 a3 = __builtin_nontemporal_load((const f32x4*)(aRow + k0 + 20));

    v16bf aF;
#pragma unroll
    for (int j = 0; j < 4; ++j) {
      aF[j]      = (__bf16)a0[j];
      aF[4 + j]  = (__bf16)a1[j];
      aF[8 + j]  = (__bf16)a2[j];
      aF[12 + j] = (__bf16)a3[j];
    }

    // B fragments from LDS (ds_load_b128, 16B aligned, conflict-free)
#pragma unroll
    for (int t = 0; t < 8; ++t) {
      bf16x8 b0 = *(const bf16x8*)(&smem[cur][t * 16 + l16][half * 8]);
      bf16x8 b1 = *(const bf16x8*)(&smem[cur][t * 16 + l16][16 + half * 8]);
      v16bf bF;
#pragma unroll
      for (int j = 0; j < 8; ++j) { bF[j] = b0[j]; bF[8 + j] = b1[j]; }
      acc[t] = __builtin_amdgcn_wmma_f32_16x16x32_bf16(false, aF, false, bF,
                                                       (short)0, acc[t],
                                                       false, false);
    }
    // fused rowsum: B = all ones
    accSum = __builtin_amdgcn_wmma_f32_16x16x32_bf16(false, aF, false, bOnes,
                                                     (short)0, accSum,
                                                     false, false);
  }

  // Normalize + store. D slot (lane l, reg r) = row (r+8*half), col l16.
#pragma unroll
  for (int r = 0; r < 8; ++r) {
    const float inv = 1.0f / accSum[r];
    const int m = mbase + r + half * 8;
#pragma unroll
    for (int t = 0; t < 8; ++t) {
      __builtin_nontemporal_store(acc[t][r] * inv,
                                  out + (size_t)m * DIMM + t * 16 + l16);
    }
  }
}

// ---------------------------------------------------------------------------
extern "C" void kernel_launch(void* const* d_in, const int* in_sizes, int n_in,
                              void* d_out, int out_size, void* d_ws,
                              size_t ws_size, hipStream_t stream) {
  const float* adj = (const float*)d_in[0];   // [8192, 8192] fp32
  const float* X   = (const float*)d_in[1];   // [8192, 128]  fp32
  const float* W   = (const float*)d_in[2];   // [128, 128]   fp32
  float* out = (float*)d_out;                 // [8192, 128]  fp32
  __bf16* xwT = (__bf16*)d_ws;                // [128, 8192]  bf16 (2 MB)

  gcn_xw_kernel<<<dim3(NN / 16, DIMM / 16), 32, 0, stream>>>(X, W, xwT);
  gcn_agg_kernel<<<dim3(NN / 64), 128, 0, stream>>>(adj, xwT, out);
}
